// Seq2Seq_54666343743960
// MI455X (gfx1250) — matrix operand order repaired and checked by reading, not compiled, not run on hardware
//
#include <hip/hip_runtime.h>
#include <hip/hip_bf16.h>
#include <stdint.h>

typedef __bf16 bf16;
typedef __attribute__((ext_vector_type(16))) __bf16 v16bf;
typedef __attribute__((ext_vector_type(8)))  __bf16 v8bf;
typedef __attribute__((ext_vector_type(8)))  float  v8f;

#define WMMA_BF16(A, B, C) \
  __builtin_amdgcn_wmma_f32_16x16x32_bf16(false, (A), false, (B), (short)0, (C), false, false)

// ---------------- constants ----------------
#define BB   64
#define SS   48
#define TT   48
#define EE   256
#define HH   512
#define VV   32000

// ---------------- fragment loaders (bf16, 16x16x32) ----------------
// A: 16x32 (MxK). lane l: m = l&15, hi = l>>4; holds K = hi*16 .. hi*16+15 (contiguous)
static __device__ __forceinline__ v16bf load_a_frag(const bf16* __restrict__ x, int ld,
                                                    int rowBase, int k0, int lane) {
  int m = lane & 15, hi = lane >> 4;
  return *(const v16bf*)(x + (size_t)(rowBase + m) * ld + k0 + hi * 16);
}
// B: 32x16 (KxN), sourced from row-major W(out,K): B[k][n] = W[colBase+n][k].
// lane l: n = l&15, hi = l>>4; holds K = hi*16 .. hi*16+15 (contiguous in W's row)
static __device__ __forceinline__ v16bf load_b_frag(const bf16* __restrict__ w, int ld,
                                                    int colBase, int k0, int lane) {
  int n = lane & 15, hi = lane >> 4;
  return *(const v16bf*)(w + (size_t)(colBase + n) * ld + k0 + hi * 16);
}

// ---------------- fused GRU step (one layer, one time step) ----------------
// gates: r = sig(x·WihR + h·WhhR + br), z likewise, n = tanh(x·WihN + bin + r*(h·WhhN + bhn))
// h' = (1-z)*n + z*h ; optional length mask (encoder); writes f32 + bf16 state (+ys slice).
// Grid: 32 blocks x 128 threads = 128 waves; wave -> one 16(row)x16(gate-col) tile strip.
__global__ __launch_bounds__(128) void gru_step_wmma(
    const bf16*  __restrict__ x,       int Kx,               // (B,Kx) bf16
    const float* __restrict__ h_in,                           // (B,H) f32
    const bf16*  __restrict__ h_in_bf,                        // (B,H) bf16
    const bf16*  __restrict__ Wih,                            // (3H,Kx) bf16
    const bf16*  __restrict__ Whh,                            // (3H,H) bf16
    const float* __restrict__ bih, const float* __restrict__ bhh, // (3H,)
    const int*   __restrict__ lens,  int t,                   // mask: keep old h where t>=lens[b]
    float*       __restrict__ h_out,                          // (B,H) f32
    bf16*        __restrict__ h_out_bf,                       // (B,H) bf16
    bf16*        __restrict__ ys_bf)                          // (B,H) bf16 slice or nullptr
{
  const int lane = threadIdx.x & 31;
  const int wave = threadIdx.x >> 5;
  const int tile = blockIdx.x * 4 + wave;   // 0..127
  const int mt = tile & 3;                  // 4 row tiles (B=64)
  const int jt = tile >> 2;                 // 32 col tiles (H=512)
  const int rowBase = mt * 16;
  const int j0 = jt * 16;

  v8f accR = {}, accZ = {}, accI = {}, accHN = {};

  // x @ Wih^T  (columns j0 / H+j0 / 2H+j0)
  for (int k0 = 0; k0 < Kx; k0 += 32) {
    v16bf a  = load_a_frag(x, Kx, rowBase, k0, lane);
    v16bf bR = load_b_frag(Wih, Kx, j0,          k0, lane);
    v16bf bZ = load_b_frag(Wih, Kx, HH + j0,     k0, lane);
    v16bf bN = load_b_frag(Wih, Kx, 2 * HH + j0, k0, lane);
    accR = WMMA_BF16(a, bR, accR);
    accZ = WMMA_BF16(a, bZ, accZ);
    accI = WMMA_BF16(a, bN, accI);
  }
  // h @ Whh^T  (r/z fuse into same accumulators; n-gate hidden part kept separate)
  for (int k0 = 0; k0 < HH; k0 += 32) {
    v16bf a  = load_a_frag(h_in_bf, HH, rowBase, k0, lane);
    v16bf bR = load_b_frag(Whh, HH, j0,          k0, lane);
    v16bf bZ = load_b_frag(Whh, HH, HH + j0,     k0, lane);
    v16bf bN = load_b_frag(Whh, HH, 2 * HH + j0, k0, lane);
    accR  = WMMA_BF16(a, bR, accR);
    accZ  = WMMA_BF16(a, bZ, accZ);
    accHN = WMMA_BF16(a, bN, accHN);
  }

  // epilogue: C layout elem v -> (m = v + 8*hi, n = lane&15)
  const int n  = lane & 15, hi = lane >> 4;
  const int col = j0 + n;
  const float br = bih[col] + bhh[col];
  const float bz = bih[HH + col] + bhh[HH + col];
  const float bi = bih[2 * HH + col];
  const float bh = bhh[2 * HH + col];
  #pragma unroll
  for (int v = 0; v < 8; ++v) {
    const int row = rowBase + v + hi * 8;
    const float hold = h_in[(size_t)row * HH + col];
    const float r  = 1.f / (1.f + __expf(-(accR[v] + br)));
    const float z  = 1.f / (1.f + __expf(-(accZ[v] + bz)));
    const float nn = tanhf(accI[v] + bi + r * (accHN[v] + bh));
    float hnew = (1.f - z) * nn + z * hold;
    if (lens && t >= lens[row]) hnew = hold;
    h_out[(size_t)row * HH + col]    = hnew;
    h_out_bf[(size_t)row * HH + col] = (bf16)hnew;
    if (ys_bf) ys_bf[(size_t)row * HH + col] = (bf16)hnew;
  }
}

// ---------------- big FC: (47*64) x 32000, K=512 ----------------
// Register-blocked: each wave computes a 16(row) x 64(col) strip = 4 N-tiles,
// reusing one A fragment across 4 WMMAs (1.25 fragment loads per WMMA).
__global__ __launch_bounds__(256) void fc_wmma(
    const bf16*  __restrict__ ys,   // (3008, 512) bf16, row = t*64 + b
    const bf16*  __restrict__ W,    // (32000, 512) bf16 (L2-resident)
    const float* __restrict__ bias, // (32000,)
    float*       __restrict__ out)  // (64, 48, 32000) f32
{
  const int lane = threadIdx.x & 31;
  const int wave = threadIdx.x >> 5;
  const int ntBase = blockIdx.x * 4;        // 500 x-blocks -> 2000 N-tiles
  const int mt = blockIdx.y * 8 + wave;     // 0..187 valid (188 M-tiles)
  if (mt >= 188) return;                    // wave-uniform exit

  const int rowBase = mt * 16, colBase = ntBase * 16;
  v8f acc0 = {}, acc1 = {}, acc2 = {}, acc3 = {};
  #pragma unroll 4
  for (int k0 = 0; k0 < HH; k0 += 32) {
    v16bf a  = load_a_frag(ys, HH, rowBase, k0, lane);
    v16bf b0 = load_b_frag(W, HH, colBase,      k0, lane);
    v16bf b1 = load_b_frag(W, HH, colBase + 16, k0, lane);
    v16bf b2 = load_b_frag(W, HH, colBase + 32, k0, lane);
    v16bf b3 = load_b_frag(W, HH, colBase + 48, k0, lane);
    acc0 = WMMA_BF16(a, b0, acc0);
    acc1 = WMMA_BF16(a, b1, acc1);
    acc2 = WMMA_BF16(a, b2, acc2);
    acc3 = WMMA_BF16(a, b3, acc3);
  }
  const int n = lane & 15, hi = lane >> 4;
  const v8f accs[4] = { acc0, acc1, acc2, acc3 };
  #pragma unroll
  for (int s = 0; s < 4; ++s) {
    const int v = colBase + s * 16 + n;
    const float bv = bias[v];
    #pragma unroll
    for (int e = 0; e < 8; ++e) {
      const int row = rowBase + e + hi * 8;   // row = t*64 + b
      const int tt = row >> 6, b = row & 63;
      out[((size_t)b * TT + (tt + 1)) * VV + v] = accs[s][e] + bv;
    }
  }
}

// ---------------- helpers ----------------
__global__ void cvt_f32_bf16(const float* __restrict__ src, bf16* __restrict__ dst, int n) {
  int i = blockIdx.x * blockDim.x + threadIdx.x;
  if (i < n) dst[i] = (bf16)src[i];
}
// dst[t*B*E + b*E + e] = bf16(emb[tok[b*stride + t]*E + e])
__global__ void gather_emb_bf16(const int* __restrict__ tok, int tokStride,
                                const float* __restrict__ emb, bf16* __restrict__ dst, int n) {
  int i = blockIdx.x * blockDim.x + threadIdx.x;
  if (i >= n) return;
  int e = i & (EE - 1);
  int b = (i >> 8) & (BB - 1);
  int t = i >> 14;
  int id = tok[b * tokStride + t];
  dst[i] = (bf16)emb[(size_t)id * EE + e];
}
__global__ void zero_f32(float* p, int n) {
  int i = blockIdx.x * blockDim.x + threadIdx.x;
  if (i < n) p[i] = 0.f;
}
__global__ void zero_bf16k(bf16* p, int n) {
  int i = blockIdx.x * blockDim.x + threadIdx.x;
  if (i < n) p[i] = (bf16)0.f;
}
__global__ void zero_out_t0(float* __restrict__ out) {
  int i = blockIdx.x * blockDim.x + threadIdx.x;   // n = 64*32000
  if (i >= BB * VV) return;
  int b = i / VV, v = i - b * VV;
  out[(size_t)b * TT * VV + v] = 0.f;
}

// ---------------- host orchestration ----------------
extern "C" void kernel_launch(void* const* d_in, const int* in_sizes, int n_in,
                              void* d_out, int out_size, void* d_ws, size_t ws_size,
                              hipStream_t stream) {
  (void)in_sizes; (void)n_in; (void)out_size; (void)ws_size;
  const int*   source   = (const int*)d_in[0];
  const int*   target   = (const int*)d_in[1];
  const int*   src_len  = (const int*)d_in[2];
  const float* enc_emb  = (const float*)d_in[3];
  const float* eWih0 = (const float*)d_in[4],  *eWhh0 = (const float*)d_in[5];
  const float* ebih0 = (const float*)d_in[6],  *ebhh0 = (const float*)d_in[7];
  const float* eWih1 = (const float*)d_in[8],  *eWhh1 = (const float*)d_in[9];
  const float* ebih1 = (const float*)d_in[10], *ebhh1 = (const float*)d_in[11];
  const float* dec_emb  = (const float*)d_in[12];
  const float* dWih0 = (const float*)d_in[13], *dWhh0 = (const float*)d_in[14];
  const float* dbih0 = (const float*)d_in[15], *dbhh0 = (const float*)d_in[16];
  const float* dWih1 = (const float*)d_in[17], *dWhh1 = (const float*)d_in[18];
  const float* dbih1 = (const float*)d_in[19], *dbhh1 = (const float*)d_in[20];
  const float* fc_W  = (const float*)d_in[21], *fc_b  = (const float*)d_in[22];
  float* out = (float*)d_out;

  // carve workspace (256B aligned)
  uintptr_t cur = (uintptr_t)d_ws;
  auto carve = [&](size_t bytes) -> void* {
    void* p = (void*)cur; cur += (bytes + 255) & ~(size_t)255; return p;
  };
  bf16* wEih0 = (bf16*)carve((size_t)3*HH*EE*2);
  bf16* wEhh0 = (bf16*)carve((size_t)3*HH*HH*2);
  bf16* wEih1 = (bf16*)carve((size_t)3*HH*HH*2);
  bf16* wEhh1 = (bf16*)carve((size_t)3*HH*HH*2);
  bf16* wDih0 = (bf16*)carve((size_t)3*HH*EE*2);
  bf16* wDhh0 = (bf16*)carve((size_t)3*HH*HH*2);
  bf16* wDih1 = (bf16*)carve((size_t)3*HH*HH*2);
  bf16* wDhh1 = (bf16*)carve((size_t)3*HH*HH*2);
  bf16* wFC   = (bf16*)carve((size_t)VV*HH*2);
  bf16* xsE   = (bf16*)carve((size_t)SS*BB*EE*2);
  bf16* xsD   = (bf16*)carve((size_t)(TT-1)*BB*EE*2);
  bf16* ys    = (bf16*)carve((size_t)(TT-1)*BB*HH*2);
  float* h0f[2] = { (float*)carve((size_t)BB*HH*4), (float*)carve((size_t)BB*HH*4) };
  float* h1f[2] = { (float*)carve((size_t)BB*HH*4), (float*)carve((size_t)BB*HH*4) };
  bf16*  h0b[2] = { (bf16*)carve((size_t)BB*HH*2),  (bf16*)carve((size_t)BB*HH*2) };
  bf16*  h1b[2] = { (bf16*)carve((size_t)BB*HH*2),  (bf16*)carve((size_t)BB*HH*2) };

  auto cvt = [&](const float* s, bf16* d, int n) {
    cvt_f32_bf16<<<(n + 255) / 256, 256, 0, stream>>>(s, d, n);
  };
  cvt(eWih0, wEih0, 3*HH*EE); cvt(eWhh0, wEhh0, 3*HH*HH);
  cvt(eWih1, wEih1, 3*HH*HH); cvt(eWhh1, wEhh1, 3*HH*HH);
  cvt(dWih0, wDih0, 3*HH*EE); cvt(dWhh0, wDhh0, 3*HH*HH);
  cvt(dWih1, wDih1, 3*HH*HH); cvt(dWhh1, wDhh1, 3*HH*HH);
  cvt(fc_W,  wFC,   VV*HH);

  { int n = SS*BB*EE;
    gather_emb_bf16<<<(n + 255)/256, 256, 0, stream>>>(source, SS, enc_emb, xsE, n); }
  { int n = (TT-1)*BB*EE;
    gather_emb_bf16<<<(n + 255)/256, 256, 0, stream>>>(target, TT, dec_emb, xsD, n); }

  { int n = BB*HH;
    zero_f32 <<<(n+255)/256, 256, 0, stream>>>(h0f[0], n);
    zero_f32 <<<(n+255)/256, 256, 0, stream>>>(h1f[0], n);
    zero_bf16k<<<(n+255)/256, 256, 0, stream>>>(h0b[0], n);
    zero_bf16k<<<(n+255)/256, 256, 0, stream>>>(h1b[0], n); }
  zero_out_t0<<<(BB*VV + 255)/256, 256, 0, stream>>>(out);

  int p0 = 0, p1 = 0;
  // encoder: 48 masked steps, 2 layers
  for (int t = 0; t < SS; ++t) {
    gru_step_wmma<<<32, 128, 0, stream>>>(xsE + (size_t)t*BB*EE, EE,
        h0f[p0], h0b[p0], wEih0, wEhh0, ebih0, ebhh0, src_len, t,
        h0f[1-p0], h0b[1-p0], (bf16*)nullptr);
    p0 ^= 1;
    gru_step_wmma<<<32, 128, 0, stream>>>(h0b[p0], HH,
        h1f[p1], h1b[p1], wEih1, wEhh1, ebih1, ebhh1, src_len, t,
        h1f[1-p1], h1b[1-p1], (bf16*)nullptr);
    p1 ^= 1;
  }
  // decoder: 47 steps, layer-1 output -> ys
  for (int t = 0; t < TT - 1; ++t) {
    gru_step_wmma<<<32, 128, 0, stream>>>(xsD + (size_t)t*BB*EE, EE,
        h0f[p0], h0b[p0], wDih0, wDhh0, dbih0, dbhh0, (const int*)nullptr, t,
        h0f[1-p0], h0b[1-p0], (bf16*)nullptr);
    p0 ^= 1;
    gru_step_wmma<<<32, 128, 0, stream>>>(h0b[p0], HH,
        h1f[p1], h1b[p1], wDih1, wDhh1, dbih1, dbhh1, (const int*)nullptr, t,
        h1f[1-p1], h1b[1-p1], ys + (size_t)t*BB*HH);
    p1 ^= 1;
  }
  // logits: 98.6 GFLOP bf16 WMMA GEMM, fc_W bf16 resident in L2,
  // 16x64 strip per wave (4 accumulators, A-fragment reused 4x)
  fc_wmma<<<dim3(500, 24), 256, 0, stream>>>(ys, wFC, fc_b, out);
}
